// AttentionLayer_80668075753671
// MI455X (gfx1250) — compile-verified, matrix operand order checked by reading, not run on hardware
//
#include <hip/hip_runtime.h>
#include <hip/hip_bf16.h>

typedef __attribute__((ext_vector_type(16))) _Float16 v16h;
typedef __attribute__((ext_vector_type(8)))  float    v8f;
typedef __attribute__((ext_vector_type(8)))  _Float16 h8;

#define B_  2
#define S_  2048
#define D_  1024
#define H_  16
#define HD_ 64

__device__ __forceinline__ int lane_id() { return threadIdx.x & 31; }

// Fragment load with wide (b128) accesses.  Works for:
//  - A fragments (16x32 f16) from a row-major tile (lane = M row), and
//  - B fragments (32x16 f16) from a TRANSPOSED tile T[n][k] (lane = N col),
// since both reduce to: lane (l&15) reads two 8-half runs at K = kh and 16+kh.
// All addresses are 16-byte aligned (ld is a multiple of 8, kh is 0 or 8).
__device__ __forceinline__ v16h frag_ld(const _Float16* src, int ld) {
  const int l  = lane_id();
  const int m  = l & 15;
  const int kh = (l >> 4) << 3;
  const _Float16* r = src + (size_t)m * ld + kh;
  const h8 lo = *(const h8*)(r);
  const h8 hi = *(const h8*)(r + 16);
  return __builtin_shufflevector(lo, hi, 0, 1, 2, 3, 4, 5, 6, 7,
                                         8, 9, 10, 11, 12, 13, 14, 15);
}

// ---------------------------------------------------------------------------
// Kernel 1: QKV projection GEMM.  X = (x @ W + b) [* 0.125 for Q], f16 output
// in head-major [B,H,S,HD] layout.  Grid: (BS/64, D/128, 3), block 256.
// ---------------------------------------------------------------------------
__global__ __launch_bounds__(256) void qkv_proj_kernel(
    const float* __restrict__ q, const float* __restrict__ k, const float* __restrict__ v,
    const float* __restrict__ Wq, const float* __restrict__ bq,
    const float* __restrict__ Wk, const float* __restrict__ bk,
    const float* __restrict__ Wv, const float* __restrict__ bv,
    _Float16* __restrict__ Xq, _Float16* __restrict__ Xk, _Float16* __restrict__ Xv)
{
  __shared__ __align__(16) _Float16 Xt[64 * 32];    // 64 M x 32 K  (4 KB), row-major
  __shared__ __align__(16) _Float16 WtT[128 * 32];  // 128 N x 32 K (8 KB), TRANSPOSED

  const int z = blockIdx.z;
  const float* x    = (z == 0) ? q  : (z == 1) ? k  : v;
  const float* W    = (z == 0) ? Wq : (z == 1) ? Wk : Wv;
  const float* bias = (z == 0) ? bq : (z == 1) ? bk : bv;
  _Float16*    X    = (z == 0) ? Xq : (z == 1) ? Xk : Xv;
  const float oscale = (z == 0) ? 0.125f : 1.0f;  // fold 1/sqrt(HD) into Q

  const int m0   = blockIdx.x * 64;
  const int n0   = blockIdx.y * 128;
  const int t    = threadIdx.x;
  const int wave = t >> 5;
  const int mt   = (wave & 3) * 16;   // wave M sub-tile within 64
  const int nt0  = (wave >> 2) * 64;  // wave N sub-tile within 128

  v8f acc[4] = {};

  for (int k0 = 0; k0 < D_; k0 += 32) {
    // Stage X tile (64x32 f32 -> f16, row-major): thread t -> row t/4, cols (t%4)*8.
    {
      const int r = t >> 2, c = (t & 3) * 8;
      const float4 a0 = *(const float4*)(x + (size_t)(m0 + r) * D_ + k0 + c);
      const float4 a1 = *(const float4*)(x + (size_t)(m0 + r) * D_ + k0 + c + 4);
      _Float16* d = Xt + r * 32 + c;
      d[0] = (_Float16)a0.x; d[1] = (_Float16)a0.y; d[2] = (_Float16)a0.z; d[3] = (_Float16)a0.w;
      d[4] = (_Float16)a1.x; d[5] = (_Float16)a1.y; d[6] = (_Float16)a1.z; d[7] = (_Float16)a1.w;
    }
    // Stage W tile (32 K x 128 N f32) TRANSPOSED to WtT[n][k] so B-fragment
    // reads are contiguous: thread t -> k-row t/8, n-cols (t%8)*16..+16.
    {
      const int r = t >> 3, c = (t & 7) * 16;
      const float* s = W + (size_t)(k0 + r) * D_ + n0 + c;
#pragma unroll
      for (int j = 0; j < 16; j += 4) {
        const float4 a = *(const float4*)(s + j);
        WtT[(c + j + 0) * 32 + r] = (_Float16)a.x;
        WtT[(c + j + 1) * 32 + r] = (_Float16)a.y;
        WtT[(c + j + 2) * 32 + r] = (_Float16)a.z;
        WtT[(c + j + 3) * 32 + r] = (_Float16)a.w;
      }
    }
    __syncthreads();

    const v16h a = frag_ld(Xt + mt * 32, 32);
#pragma unroll
    for (int nsub = 0; nsub < 4; ++nsub) {
      const v16h bm = frag_ld(WtT + (nt0 + nsub * 16) * 32, 32);
      acc[nsub] = __builtin_amdgcn_wmma_f32_16x16x32_f16(
          false, a, false, bm, (short)0, acc[nsub], false, false);
    }
    __syncthreads();
  }

  // Epilogue: bias + scale, scatter to head-major f16 workspace.
  const int l     = lane_id();
  const int nlane = l & 15;
  const int mb    = (l >> 4) << 3;
#pragma unroll
  for (int nsub = 0; nsub < 4; ++nsub) {
    const int n  = n0 + nt0 + nsub * 16 + nlane;
    const float bb = bias[n];
    const int h  = n >> 6, hd = n & 63;
#pragma unroll
    for (int i = 0; i < 8; ++i) {
      const int m = m0 + mt + mb + i;
      const int bb_idx = m >> 11;            // batch
      const int ss     = m & (S_ - 1);       // sequence pos
      X[(((size_t)(bb_idx * H_ + h)) * S_ + ss) * HD_ + hd] =
          (_Float16)((acc[nsub][i] + bb) * oscale);
    }
  }
}

// ---------------------------------------------------------------------------
// Kernel 2: flash attention over head-major f16 Q/K/V.
// Grid: (S/128, B*H), block 256 (8 waves; each wave owns 16 q-rows).
// ---------------------------------------------------------------------------
__global__ __launch_bounds__(256) void attn_kernel(
    const _Float16* __restrict__ Xq, const _Float16* __restrict__ Xk,
    const _Float16* __restrict__ Xv, float* __restrict__ out)
{
  __shared__ __align__(16) _Float16 Kt[32 * HD_];    // 4 KB, row-major [k][d]
  __shared__ __align__(16) _Float16 VtT[HD_ * 32];   // 4 KB, TRANSPOSED [d][k]
  __shared__ __align__(16) _Float16 Pt[8][16 * 32];  // 8 KB, wave-private P tiles

  const int bh   = blockIdx.y;           // 0..B*H-1
  const int b    = bh >> 4;
  const int h    = bh & 15;
  const int t    = threadIdx.x;
  const int wave = t >> 5;
  const int l    = lane_id();
  const int q0   = blockIdx.x * 128 + wave * 16;

  const size_t head_base = (size_t)bh * S_ * HD_;

  // Q fragments for this wave's 16 rows (pre-scaled by 1/sqrt(HD)); two
  // global_load_b128 per lane each.
  const v16h aq0 = frag_ld(Xq + head_base + (size_t)q0 * HD_, HD_);
  const v16h aq1 = frag_ld(Xq + head_base + (size_t)q0 * HD_ + 32, HD_);

  v8f   o[4] = {};                       // running 16x64 output accumulator
  float mrun[8], lrun[8];
#pragma unroll
  for (int i = 0; i < 8; ++i) { mrun[i] = -1e30f; lrun[i] = 0.0f; }

  const int nlane = l & 15;
  const int mb    = (l >> 4) << 3;
  _Float16* Pw = &Pt[wave][0];

  for (int kc = 0; kc < S_; kc += 32) {
    // Cooperative K/V chunk load (32x64 f16 each): thread t -> row t/8, 8 halves.
    // K stays row-major; V goes in transposed so P@V B-fragments load wide.
    {
      const int r  = t >> 3, c8 = (t & 7) * 8;
      const size_t off = head_base + (size_t)(kc + r) * HD_ + c8;
      *(h8*)(Kt + r * HD_ + c8) = *(const h8*)(Xk + off);
      const h8 vv = *(const h8*)(Xv + off);
#pragma unroll
      for (int j = 0; j < 8; ++j) VtT[(c8 + j) * 32 + r] = vv[j];
      if (kc + 32 < S_) {                // prefetch next chunk -> global_prefetch_b8
        __builtin_prefetch(Xk + off + (size_t)32 * HD_, 0, 1);
        __builtin_prefetch(Xv + off + (size_t)32 * HD_, 0, 1);
      }
    }
    __syncthreads();

    // Scores: two 16x16 tiles over this chunk's 32 k-positions.
    // B = K^T, so the B-fragment loads with A addressing from row-major K.
    v8f s0 = {}, s1 = {};
    {
      const v16h bk0 = frag_ld(Kt, HD_);            // k 0..15, d 0..31
      const v16h bk1 = frag_ld(Kt + 32, HD_);       // k 0..15, d 32..63
      s0 = __builtin_amdgcn_wmma_f32_16x16x32_f16(false, aq0, false, bk0, (short)0, s0, false, false);
      s0 = __builtin_amdgcn_wmma_f32_16x16x32_f16(false, aq1, false, bk1, (short)0, s0, false, false);
      const v16h bk2 = frag_ld(Kt + 16 * HD_, HD_);      // k 16..31, d 0..31
      const v16h bk3 = frag_ld(Kt + 16 * HD_ + 32, HD_); // k 16..31, d 32..63
      s1 = __builtin_amdgcn_wmma_f32_16x16x32_f16(false, aq0, false, bk2, (short)0, s1, false, false);
      s1 = __builtin_amdgcn_wmma_f32_16x16x32_f16(false, aq1, false, bk3, (short)0, s1, false, false);
    }

    // Online softmax (rows live across 16-lane halves; shfl_xor masks <= 8
    // never cross bit 4, so each half reduces independently in wave32).
#pragma unroll
    for (int i = 0; i < 8; ++i) {
      float rmax = fmaxf(s0[i], s1[i]);
#pragma unroll
      for (int xm = 8; xm >= 1; xm >>= 1) rmax = fmaxf(rmax, __shfl_xor(rmax, xm, 32));
      const float mnew = fmaxf(mrun[i], rmax);
      const float corr = __expf(mrun[i] - mnew);
      mrun[i] = mnew;
      const float p0 = __expf(s0[i] - mnew);
      const float p1 = __expf(s1[i] - mnew);
      float psum = p0 + p1;
#pragma unroll
      for (int xm = 8; xm >= 1; xm >>= 1) psum += __shfl_xor(psum, xm, 32);
      lrun[i] = lrun[i] * corr + psum;
#pragma unroll
      for (int nsub = 0; nsub < 4; ++nsub) o[nsub][i] *= corr;
      const int m = mb + i;
      Pw[m * 32 + nlane]      = (_Float16)p0;   // C-frag -> A-frag relayout via LDS
      Pw[m * 32 + 16 + nlane] = (_Float16)p1;
    }

    // out += P @ V  (A = 16x32 probs from row-major Pw, B from transposed VtT).
    const v16h ap = frag_ld(Pw, 32);
#pragma unroll
    for (int nsub = 0; nsub < 4; ++nsub) {
      const v16h bvf = frag_ld(VtT + (nsub * 16) * 32, 32);
      o[nsub] = __builtin_amdgcn_wmma_f32_16x16x32_f16(
          false, ap, false, bvf, (short)0, o[nsub], false, false);
    }
    __syncthreads();
  }

  // Epilogue: normalize and write merged-head f32 output [B,S,D].
#pragma unroll
  for (int nsub = 0; nsub < 4; ++nsub) {
#pragma unroll
    for (int i = 0; i < 8; ++i) {
      const int s = q0 + mb + i;
      const int d = h * HD_ + nsub * 16 + nlane;
      out[((size_t)b * S_ + s) * D_ + d] = o[nsub][i] / lrun[i];
    }
  }
}

extern "C" void kernel_launch(void* const* d_in, const int* in_sizes, int n_in,
                              void* d_out, int out_size, void* d_ws, size_t ws_size,
                              hipStream_t stream) {
  (void)in_sizes; (void)n_in; (void)out_size; (void)ws_size;
  const float* q  = (const float*)d_in[0];
  const float* k  = (const float*)d_in[1];
  const float* v  = (const float*)d_in[2];
  const float* Wq = (const float*)d_in[3];
  const float* bq = (const float*)d_in[4];
  const float* Wk = (const float*)d_in[5];
  const float* bk = (const float*)d_in[6];
  const float* Wv = (const float*)d_in[7];
  const float* bv = (const float*)d_in[8];
  float* out = (float*)d_out;

  const size_t nX = (size_t)B_ * H_ * S_ * HD_;   // 4M elems -> 8 MB f16 each
  _Float16* Xq = (_Float16*)d_ws;
  _Float16* Xk = Xq + nX;
  _Float16* Xv = Xk + nX;

  dim3 gp(B_ * S_ / 64, D_ / 128, 3);
  qkv_proj_kernel<<<gp, 256, 0, stream>>>(q, k, v, Wq, bq, Wk, bk, Wv, bv, Xq, Xk, Xv);

  dim3 ga(S_ / 128, B_ * H_);
  attn_kernel<<<ga, 256, 0, stream>>>(Xq, Xk, Xv, out);
}
